// AttentionHead_71081708748919
// MI455X (gfx1250) — matrix operand order repaired and checked by reading, not compiled, not run on hardware
//
#include <hip/hip_runtime.h>
#include <hip/hip_bf16.h>

#define BB 8
#define NN 2048
#define DD 768
#define HH 64
#define CH 64   // keys per chunk in the attention kernel

typedef __attribute__((ext_vector_type(16))) _Float16 v16h;
typedef __attribute__((ext_vector_type(8)))  _Float16 v8h;
typedef __attribute__((ext_vector_type(8)))  float    v8f;
typedef __attribute__((ext_vector_type(4)))  float    v4f;
typedef __attribute__((ext_vector_type(4)))  int      v4i;

#if defined(__AMDGCN__) && __has_builtin(__builtin_amdgcn_global_load_async_to_lds_b128) && __has_builtin(__builtin_amdgcn_s_wait_asynccnt)
#define ASYNC_OK 1
#else
#define ASYNC_OK 0
#endif

#if ASYNC_OK
// builtin signature: void(v4i addrspace(1)*, v4i addrspace(3)*, Imm offset, Imm cpol)
typedef __attribute__((address_space(1))) v4i glb_v4i;
typedef __attribute__((address_space(3))) v4i lds_v4i;
#endif

static __device__ __forceinline__ v8f wmma_f16(v16h a, v16h b, v8f c) {
    // D = A(16x32 f16) * B(32x16 f16) + C(16x16 f32)
    return __builtin_amdgcn_wmma_f32_16x16x32_f16(
        /*neg_a=*/false, a, /*neg_b=*/false, b,
        /*c_mod=*/(short)0, c, /*reuse_a=*/false, /*reuse_b=*/false);
}

// A-fragment (16x32, f16) built from an f32 row. Layout: lanes 0-15 hold
// K 0..15, lanes 16-31 hold K 16..31 of row m = lane%16.
static __device__ __forceinline__ v16h load_a_frag_f32(const float* __restrict__ row,
                                                       int k0, int sel) {
    const v4f* p0 = (const v4f*)(row + k0 + sel * 16);
    const v4f* p1 = (const v4f*)(row + k0 + 8 + sel * 16);
    v4f x0 = p0[0], x1 = p0[1], x2 = p1[0], x3 = p1[1];
    v16h a;
#pragma unroll
    for (int i = 0; i < 4; ++i) {
        a[i]      = (_Float16)x0[i];
        a[4 + i]  = (_Float16)x1[i];
        a[8 + i]  = (_Float16)x2[i];
        a[12 + i] = (_Float16)x3[i];
    }
    return a;
}

// A-fragment from an f16 row (same contiguous K layout).
static __device__ __forceinline__ v16h load_a_frag_h16(const _Float16* row, int sel) {
    v8h lo = *(const v8h*)(row + sel * 16);
    v8h hi = *(const v8h*)(row + 8 + sel * 16);
    v16h a;
#pragma unroll
    for (int i = 0; i < 8; ++i) { a[i] = lo[i]; a[8 + i] = hi[i]; }
    return a;
}

// B-fragment (32x16, f16): lane holds column n = lane%16; lanes 0-15 hold
// K 0..15, lanes 16-31 hold K 16..31 -> one contiguous 16-half segment.
static __device__ __forceinline__ v16h load_b_frag(const _Float16* p /*16B aligned*/) {
    v8h lo = *(const v8h*)p;
    v8h hi = *(const v8h*)(p + 8);
    v16h b;
#pragma unroll
    for (int i = 0; i < 8; ++i) { b[i] = lo[i]; b[8 + i] = hi[i]; }
    return b;
}

// ---------------------------------------------------------------------------
// QKV projection: q/k/v = f16(x @ W + b). One block = 4 waves = 64 rows.
// grid = ((B*N)/64, 3); blockIdx.y selects q/k/v.
// ---------------------------------------------------------------------------
__global__ __launch_bounds__(128)
void proj_qkv_kernel(const float* __restrict__ x,
                     const float* __restrict__ Wq, const float* __restrict__ bq,
                     const float* __restrict__ Wk, const float* __restrict__ bk,
                     const float* __restrict__ Wv, const float* __restrict__ bv,
                     _Float16* __restrict__ qo, _Float16* __restrict__ ko,
                     _Float16* __restrict__ vo) {
    const float* W;
    const float* bias;
    _Float16* out;
    if (blockIdx.y == 0)      { W = Wq; bias = bq; out = qo; }
    else if (blockIdx.y == 1) { W = Wk; bias = bk; out = ko; }
    else                      { W = Wv; bias = bv; out = vo; }

    // W chunk transposed: Wt[col][kk], padded row stride 40 halves (80B).
    __shared__ _Float16 Wt[64][40];

    const int tid = threadIdx.x;
    const int wave = tid >> 5, lane = tid & 31;
    const int n = lane & 15, sel = lane >> 4;
    const long rowBase = (long)blockIdx.x * 64 + wave * 16;
    const float* xrow = x + (rowBase + n) * (long)DD;

    // staging assignment: thread -> (chunk row d, 16-col quarter cb); the
    // loads are contiguous float4s (coalesced), the transpose happens on the
    // LDS-store side.
    const int sd = tid >> 2;          // 0..31
    const int scb = (tid & 3) * 16;   // 0/16/32/48

    v8f acc[4];
#pragma unroll
    for (int t = 0; t < 4; ++t)
#pragma unroll
        for (int i = 0; i < 8; ++i) acc[t][i] = 0.0f;

    for (int k0 = 0; k0 < DD; k0 += 32) {
        __syncthreads();
        {   // cooperative stage of W[k0..k0+31][0..63] -> Wt (f16, transposed)
            const v4f* src = (const v4f*)(W + (long)(k0 + sd) * HH + scb);
            v4f w0 = src[0], w1 = src[1], w2 = src[2], w3 = src[3];
#pragma unroll
            for (int i = 0; i < 4; ++i) {
                Wt[scb + i][sd]      = (_Float16)w0[i];
                Wt[scb + 4 + i][sd]  = (_Float16)w1[i];
                Wt[scb + 8 + i][sd]  = (_Float16)w2[i];
                Wt[scb + 12 + i][sd] = (_Float16)w3[i];
            }
            if (k0 + 32 < DD)   // prefetch next W chunk
                __builtin_prefetch(W + (long)(k0 + 32 + sd) * HH + scb, 0, 0);
        }
        __syncthreads();

        v16h a = load_a_frag_f32(xrow, k0, sel);
#pragma unroll
        for (int t = 0; t < 4; ++t)
            acc[t] = wmma_f16(a, load_b_frag(&Wt[t * 16 + n][sel * 16]), acc[t]);
    }

#pragma unroll
    for (int t = 0; t < 4; ++t) {
        int colo = t * 16 + n;
        float bb = bias[colo];
#pragma unroll
        for (int r = 0; r < 8; ++r)
            out[(rowBase + r + 8 * sel) * HH + colo] = (_Float16)(acc[t][r] + bb);
    }
}

// ---------------------------------------------------------------------------
// Fused flash attention: each wave owns a 16-row query tile, streams K/V in
// 64-key chunks staged in LDS (K via async-to-LDS DMA, V transposed through
// VGPRs). grid = (N/64, B), block = 128 (4 waves).
// ---------------------------------------------------------------------------
__global__ __launch_bounds__(128)
void attn_kernel(const _Float16* __restrict__ q, const _Float16* __restrict__ k,
                 const _Float16* __restrict__ v, const int* __restrict__ mask,
                 float* __restrict__ out) {
    __shared__ _Float16 Ksh[CH][72];      // 64 keys x 64 dims (stride 144B)
    __shared__ _Float16 Vt[HH][72];       // transposed: [dim][key] (stride 144B)
    __shared__ _Float16 Pscr[4][16][72];  // per-wave P tile (16 x 64, stride 144B)

    const int tid = threadIdx.x;
    const int wave = tid >> 5, lane = tid & 31;
    const int n = lane & 15, sel = lane >> 4;
    const int b = blockIdx.y;
    const long qBase = (long)b * NN + (long)blockIdx.x * 64 + wave * 16;

    // Q A-fragments for dims 0-31 and 32-63 (held in registers all kernel).
    const _Float16* qrow = q + (qBase + n) * HH;
    v16h aq0 = load_a_frag_h16(qrow + 0, sel);
    v16h aq1 = load_a_frag_h16(qrow + 32, sel);

    const float NEG_BIG = -1.0e30f;
    const float scale = 0.125f;                 // 1/sqrt(64)
    const float LOG2E = 1.44269504088896340736f;

    float m_i[8], l_i[8];
    v8f accO[4];
#pragma unroll
    for (int r = 0; r < 8; ++r) { m_i[r] = NEG_BIG; l_i[r] = 0.0f; }
#pragma unroll
    for (int t = 0; t < 4; ++t)
#pragma unroll
        for (int i = 0; i < 8; ++i) accO[t][i] = 0.0f;

    // staging assignment: thread -> (key, 32-dim half)
    const int skey = tid >> 1, spart = tid & 1;

    for (int kc = 0; kc < NN; kc += CH) {
        __syncthreads();
        {
            const _Float16* ks = k + ((long)b * NN + kc + skey) * HH + spart * 32;
            _Float16* kl = &Ksh[skey][spart * 32];
#if ASYNC_OK
            // direct DMA: 64B per thread, no VGPR round trip (ASYNCcnt-tracked)
            __builtin_amdgcn_global_load_async_to_lds_b128((glb_v4i*)ks, (lds_v4i*)kl, 0, 0);
            __builtin_amdgcn_global_load_async_to_lds_b128((glb_v4i*)ks, (lds_v4i*)kl, 16, 0);
            __builtin_amdgcn_global_load_async_to_lds_b128((glb_v4i*)ks, (lds_v4i*)kl, 32, 0);
            __builtin_amdgcn_global_load_async_to_lds_b128((glb_v4i*)ks, (lds_v4i*)kl, 48, 0);
#else
            {
                const v8h* s = (const v8h*)ks;
                v8h* d = (v8h*)kl;
                d[0] = s[0]; d[1] = s[1]; d[2] = s[2]; d[3] = s[3];
            }
#endif
            // V staged transposed (needs VGPR pass)
            const _Float16* vs = v + ((long)b * NN + kc + skey) * HH + spart * 32;
            v8h t0 = *(const v8h*)(vs + 0);
            v8h t1 = *(const v8h*)(vs + 8);
            v8h t2 = *(const v8h*)(vs + 16);
            v8h t3 = *(const v8h*)(vs + 24);
#pragma unroll
            for (int i = 0; i < 8; ++i) {
                Vt[spart * 32 + i][skey]      = t0[i];
                Vt[spart * 32 + 8 + i][skey]  = t1[i];
                Vt[spart * 32 + 16 + i][skey] = t2[i];
                Vt[spart * 32 + 24 + i][skey] = t3[i];
            }
            if (kc + CH < NN) {   // prefetch next chunk
                __builtin_prefetch(ks + (long)CH * HH, 0, 0);
                __builtin_prefetch(vs + (long)CH * HH, 0, 0);
            }
        }
#if ASYNC_OK
        __builtin_amdgcn_s_wait_asynccnt(0);
#endif
        __syncthreads();

        // S tiles: 4 x (16 queries x 16 keys), K-dim = 64 (2 WMMAs each)
        v8f s[4];
#pragma unroll
        for (int j = 0; j < 4; ++j) {
#pragma unroll
            for (int i = 0; i < 8; ++i) s[j][i] = 0.0f;
            s[j] = wmma_f16(aq0, load_b_frag(&Ksh[j * 16 + n][0  + sel * 16]), s[j]);
            s[j] = wmma_f16(aq1, load_b_frag(&Ksh[j * 16 + n][32 + sel * 16]), s[j]);
        }

        // scale + mask (column == this lane's key)
        int mk[4];
#pragma unroll
        for (int j = 0; j < 4; ++j) mk[j] = mask[(long)b * NN + kc + j * 16 + n];

        float rmax[8];
#pragma unroll
        for (int r = 0; r < 8; ++r) {
            float mx = NEG_BIG;
#pragma unroll
            for (int j = 0; j < 4; ++j) {
                float a0 = mk[j] ? s[j][r] * scale : NEG_BIG;
                s[j][r] = a0;
                mx = fmaxf(mx, a0);
            }
            rmax[r] = mx;
        }
        // row max across the 16 lanes of each half-wave
#pragma unroll
        for (int off = 1; off < 16; off <<= 1)
#pragma unroll
            for (int r = 0; r < 8; ++r)
                rmax[r] = fmaxf(rmax[r], __shfl_xor(rmax[r], off, 32));

        float rsum[8];
#pragma unroll
        for (int r = 0; r < 8; ++r) {
            float mn = fmaxf(m_i[r], rmax[r]);
            float al = exp2f((m_i[r] - mn) * LOG2E);
            m_i[r] = mn;
            float acc = 0.0f;
#pragma unroll
            for (int j = 0; j < 4; ++j) {
                float p = exp2f((s[j][r] - mn) * LOG2E);
                acc += p;
                // C-layout -> row-major P scratch (row = r + 8*sel)
                Pscr[wave][r + 8 * sel][j * 16 + n] = (_Float16)p;
            }
            rsum[r] = acc;
            l_i[r] *= al;
#pragma unroll
            for (int t = 0; t < 4; ++t) accO[t][r] *= al;
        }
#pragma unroll
        for (int off = 1; off < 16; off <<= 1)
#pragma unroll
            for (int r = 0; r < 8; ++r)
                rsum[r] += __shfl_xor(rsum[r], off, 32);
#pragma unroll
        for (int r = 0; r < 8; ++r) l_i[r] += rsum[r];

        // P (16x64) as two A-fragments; V^T rows give contiguous B-fragments.
        v16h ap0 = load_a_frag_h16(&Pscr[wave][n][0], sel);
        v16h ap1 = load_a_frag_h16(&Pscr[wave][n][32], sel);
#pragma unroll
        for (int t = 0; t < 4; ++t) {
            accO[t] = wmma_f16(ap0, load_b_frag(&Vt[t * 16 + n][0  + sel * 16]), accO[t]);
            accO[t] = wmma_f16(ap1, load_b_frag(&Vt[t * 16 + n][32 + sel * 16]), accO[t]);
        }
    }

    // normalize and store f32 output
#pragma unroll
    for (int t = 0; t < 4; ++t) {
        int colo = t * 16 + n;
#pragma unroll
        for (int r = 0; r < 8; ++r) {
            float denom = l_i[r];
            float val = (denom > 0.0f) ? accO[t][r] / denom : 0.0f;
            out[(qBase + r + 8 * sel) * HH + colo] = val;
        }
    }
}

extern "C" void kernel_launch(void* const* d_in, const int* in_sizes, int n_in,
                              void* d_out, int out_size, void* d_ws, size_t ws_size,
                              hipStream_t stream) {
    const float* x    = (const float*)d_in[0];
    const int*   mask = (const int*)d_in[1];
    const float* Wq   = (const float*)d_in[2];
    const float* bq   = (const float*)d_in[3];
    const float* Wk   = (const float*)d_in[4];
    const float* bk   = (const float*)d_in[5];
    const float* Wv   = (const float*)d_in[6];
    const float* bv   = (const float*)d_in[7];
    float* out = (float*)d_out;

    // workspace: q/k/v in f16, each B*N*H halves (2 MB) -> 6 MB total
    const size_t perTensor = (size_t)BB * NN * HH;
    _Float16* qh = (_Float16*)d_ws;
    _Float16* kh = qh + perTensor;
    _Float16* vh = kh + perTensor;

    dim3 gProj((BB * NN) / 64, 3);
    proj_qkv_kernel<<<gProj, 128, 0, stream>>>(x, Wq, bq, Wk, bk, Wv, bv, qh, kh, vh);

    dim3 gAttn(NN / 64, BB);
    attn_kernel<<<gAttn, 128, 0, stream>>>(qh, kh, vh, mask, out);
}
